// simple_PyG_gcn_10720238371545
// MI455X (gfx1250) — matrix operand order repaired and checked by reading, not compiled, not run on hardware
//
#include <hip/hip_runtime.h>

typedef __attribute__((ext_vector_type(16))) __bf16 v16bf;
typedef __attribute__((ext_vector_type(8)))  __bf16 v8bf;
typedef __attribute__((ext_vector_type(8)))  float  v8f;
typedef __attribute__((ext_vector_type(4)))  float  v4f;

#define GCN_N 8192

// ---------------------------------------------------------------------------
// adj_gemm: C[M][N] = opt_relu( adj_f32[M][8192] * B_bf16[8192][N] )
// Block = 256 threads = 8 waves: 2 M-strips of 16 rows x 4 waves splitting N.
// Grid = 256 blocks (M=32 each) to spread across all WGPs / maximize MLP.
// A (adj, fp32, NT) and B (bf16) slabs both staged to LDS with double-buffered
// GLOBAL_LOAD_ASYNC_TO_LDS_B128 (ASYNCcnt), overlapped with WMMA compute.
// ---------------------------------------------------------------------------
template<int NTILES, bool RELU, typename OutT>
__global__ __launch_bounds__(256) void adj_gemm_kernel(
    const float* __restrict__ adj, const __bf16* __restrict__ B,
    OutT* __restrict__ C)
{
  constexpr int N = NTILES * 16;
  constexpr int BPITCH = N + 8;           // bf16 elems; 16B-aligned rows
  constexpr int APITCH = 36;              // f32 elems (32 + 4 pad)
  constexpr int K = GCN_N;
  constexpr int BSLAB = 32 * BPITCH;
  constexpr int ASLAB = 32 * APITCH;
  __shared__ __bf16 Bs[2 * BSLAB];
  __shared__ float  As[2 * ASLAB];

  const int wave = threadIdx.x >> 5;
  const int lane = threadIdx.x & 31;
  const int half = lane >> 4;
  const int mrow = lane & 15;
  constexpr int TPW = NTILES / 4;         // N-tiles per wave
  const int strip = wave >> 2;            // 0..1 (M strip)
  const int tbase = (wave & 3) * TPW;     // first N-tile of this wave
  const int blockM = blockIdx.x * 32;

  v8f acc[TPW] = {};

  // --- B slab: 32 rows x N bf16, 16B chunks round-robined over 256 threads
  constexpr int BCPR = N / 8;             // chunks per row
  constexpr int BPT = (32 * BCPR) / 256;  // chunks per thread
  auto issue_B = [&](int k0, int buf) {
#pragma unroll
    for (int j = 0; j < BPT; ++j) {
      const int c = (int)threadIdx.x + 256 * j;
      const int row = c / BCPR;
      const int col = (c % BCPR) * 8;
      unsigned int lds =
          (unsigned int)(size_t)&Bs[buf * BSLAB + row * BPITCH + col];
      unsigned long long ga =
          (unsigned long long)(size_t)&B[(size_t)(k0 + row) * N + col];
      asm volatile("global_load_async_to_lds_b128 %0, %1, off"
                   :: "v"(lds), "v"(ga) : "memory");
    }
  };
  // --- A slab: 32 rows x 32 f32 of adj, exactly one 16B chunk per thread.
  //     Non-temporal: adj is a single-use 256MB stream, keep L2 for B.
  auto issue_A = [&](int k0, int buf) {
    const int row = (int)threadIdx.x >> 3;
    const int col = ((int)threadIdx.x & 7) * 4;
    unsigned int lds =
        (unsigned int)(size_t)&As[buf * ASLAB + row * APITCH + col];
    const float* gp = &adj[(size_t)(blockM + row) * K + k0 + col];
    __builtin_prefetch(gp + 256, 0, 1);   // 8 K-steps (1 KB) ahead per row
    unsigned long long ga = (unsigned long long)(size_t)gp;
    asm volatile("global_load_async_to_lds_b128 %0, %1, off th:TH_LOAD_NT"
                 :: "v"(lds), "v"(ga) : "memory");
  };

  // Prologue: stage slab 0 into buffer 0.
  issue_A(0, 0);
  issue_B(0, 0);
  asm volatile("s_wait_asynccnt 0x0" ::: "memory");
  __syncthreads();

  for (int k0 = 0; k0 < K; k0 += 32) {
    const int cur = (k0 >> 5) & 1;
    if (k0 + 32 < K) {                     // overlap next slab with compute
      issue_A(k0 + 32, cur ^ 1);
      issue_B(k0 + 32, cur ^ 1);
    }

    // A fragment from LDS (fp32 -> bf16); lanes 0-15: K{0..7,16..23},
    // lanes 16-31: K{8..15,24..31} per the 16-bit A-matrix layout.
    const float* ar = &As[cur * ASLAB + (strip * 16 + mrow) * APITCH];
    v4f a0 = *(const v4f*)(ar + half * 8);
    v4f a1 = *(const v4f*)(ar + half * 8 + 4);
    v4f a2 = *(const v4f*)(ar + 16 + half * 8);
    v4f a3 = *(const v4f*)(ar + 16 + half * 8 + 4);
    v16bf a;
#pragma unroll
    for (int i = 0; i < 4; ++i) {
      a[i]      = (__bf16)a0[i];
      a[4 + i]  = (__bf16)a1[i];
      a[8 + i]  = (__bf16)a2[i];
      a[12 + i] = (__bf16)a3[i];
    }
#pragma unroll
    for (int t = 0; t < TPW; ++t) {
      // B fragment: lane = N column (mrow), K{0..7,16..23} / K{8..15,24..31} per half
      const __bf16* bc = &Bs[cur * BSLAB + (tbase + t) * 16 + mrow];
      v16bf b;
#pragma unroll
      for (int i = 0; i < 8; ++i) { b[i] = bc[(half * 8 + i) * BPITCH]; b[8 + i] = bc[(16 + half * 8 + i) * BPITCH]; }
      acc[t] = __builtin_amdgcn_wmma_f32_16x16x32_bf16(
          false, a, false, b, (short)0, acc[t], false, false);
    }

    asm volatile("s_wait_asynccnt 0x0" ::: "memory");  // next slabs landed
    __syncthreads();
  }

#pragma unroll
  for (int t = 0; t < TPW; ++t) {
#pragma unroll
    for (int r = 0; r < 8; ++r) {
      float v = acc[t][r];
      if (RELU) v = fmaxf(v, 0.f);
      const int m = blockM + strip * 16 + half * 8 + r;
      const int n = (tbase + t) * 16 + mrow;
      C[(size_t)m * N + n] = (OutT)v;
    }
  }
}

// Emit the hot kernels first in .text (visible in disasm snippet).
template __global__ void adj_gemm_kernel<16, true, __bf16>(
    const float* __restrict__, const __bf16* __restrict__, __bf16* __restrict__);
template __global__ void adj_gemm_kernel<4, false, float>(
    const float* __restrict__, const __bf16* __restrict__, float* __restrict__);

// ---------------------------------------------------------------------------
// small_gemm: C_bf16[M][N] = A[M][256] * Wb_bf16[256][N]  (N = NTILES*16)
// ---------------------------------------------------------------------------
template<int NTILES, typename AT>
__global__ __launch_bounds__(256) void small_gemm_kernel(
    const AT* __restrict__ A, const __bf16* __restrict__ W,
    __bf16* __restrict__ C)
{
  constexpr int N = NTILES * 16;
  constexpr int K = 256;
  const int wave = threadIdx.x >> 5;
  const int lane = threadIdx.x & 31;
  const int half = lane >> 4;
  const int mrow = lane & 15;
  const int waveM = blockIdx.x * 128 + wave * 16;

  v8f acc[NTILES] = {};
  const AT* Arow = A + (size_t)(waveM + mrow) * K;

  for (int k0 = 0; k0 < K; k0 += 32) {
    v16bf a;
    if constexpr (__is_same(AT, float)) {
      v4f a0 = *(const v4f*)(Arow + k0 + half * 8);
      v4f a1 = *(const v4f*)(Arow + k0 + half * 8 + 4);
      v4f a2 = *(const v4f*)(Arow + k0 + 16 + half * 8);
      v4f a3 = *(const v4f*)(Arow + k0 + 16 + half * 8 + 4);
#pragma unroll
      for (int i = 0; i < 4; ++i) {
        a[i]      = (__bf16)a0[i];
        a[4 + i]  = (__bf16)a1[i];
        a[8 + i]  = (__bf16)a2[i];
        a[12 + i] = (__bf16)a3[i];
      }
    } else {
      v8bf alo = *(const v8bf*)(Arow + k0 + half * 8);
      v8bf ahi = *(const v8bf*)(Arow + k0 + 16 + half * 8);
#pragma unroll
      for (int i = 0; i < 8; ++i) { a[i] = alo[i]; a[8 + i] = ahi[i]; }
    }
#pragma unroll
    for (int t = 0; t < NTILES; ++t) {
      const __bf16* wc = W + (size_t)k0 * N + t * 16 + mrow;
      v16bf b;
#pragma unroll
      for (int j = 0; j < 8; ++j) { b[j] = wc[(size_t)(half * 8 + j) * N]; b[8 + j] = wc[(size_t)(16 + half * 8 + j) * N]; }
      acc[t] = __builtin_amdgcn_wmma_f32_16x16x32_bf16(
          false, a, false, b, (short)0, acc[t], false, false);
    }
  }
#pragma unroll
  for (int t = 0; t < NTILES; ++t) {
#pragma unroll
    for (int r = 0; r < 8; ++r) {
      const int m = waveM + half * 8 + r;
      const int n = t * 16 + mrow;
      C[(size_t)m * N + n] = (__bf16)acc[t][r];
    }
  }
}

template __global__ void small_gemm_kernel<16, float>(
    const float* __restrict__, const __bf16* __restrict__, __bf16* __restrict__);
template __global__ void small_gemm_kernel<4, __bf16>(
    const __bf16* __restrict__, const __bf16* __restrict__, __bf16* __restrict__);

// ---------------------------------------------------------------------------
// f32 -> bf16 bulk convert (for W1/W2; tiny, one-time per launch)
// ---------------------------------------------------------------------------
__global__ __launch_bounds__(256) void cvt_bf16_kernel(
    const float* __restrict__ src, __bf16* __restrict__ dst)
{
  const int i = (blockIdx.x * 256 + threadIdx.x) * 4;
  v4f v = *(const v4f*)(src + i);
#pragma unroll
  for (int j = 0; j < 4; ++j) dst[i + j] = (__bf16)v[j];
}

// ---------------------------------------------------------------------------
// log_softmax over 64 classes per row
// ---------------------------------------------------------------------------
__global__ __launch_bounds__(64) void logsoftmax_kernel(
    const float* __restrict__ z, float* __restrict__ out)
{
  __shared__ float red[64];
  const int row = blockIdx.x;
  const int c = threadIdx.x;
  const float v = z[(size_t)row * 64 + c];

  red[c] = v;
  __syncthreads();
#pragma unroll
  for (int s = 32; s > 0; s >>= 1) {
    if (c < s) red[c] = fmaxf(red[c], red[c + s]);
    __syncthreads();
  }
  const float m = red[0];
  __syncthreads();

  red[c] = __expf(v - m);
  __syncthreads();
#pragma unroll
  for (int s = 32; s > 0; s >>= 1) {
    if (c < s) red[c] += red[c + s];
    __syncthreads();
  }
  const float lse = __logf(red[0]) + m;
  out[(size_t)row * 64 + c] = v - lse;
}

// ---------------------------------------------------------------------------
extern "C" void kernel_launch(void* const* d_in, const int* in_sizes, int n_in,
                              void* d_out, int out_size, void* d_ws, size_t ws_size,
                              hipStream_t stream) {
  const float* adj = (const float*)d_in[0];   // [8192][8192]
  const float* x   = (const float*)d_in[1];   // [8192][256]
  const float* W1  = (const float*)d_in[2];   // [256][256]
  const float* W2  = (const float*)d_in[3];   // [256][64]
  float* out = (float*)d_out;                 // [8192][64]

  char* ws = (char*)d_ws;
  __bf16* XW     = (__bf16*)(ws);                        // 4 MB
  __bf16* H      = (__bf16*)(ws + 4u * 1024 * 1024);     // 4 MB
  __bf16* HW2    = (__bf16*)(ws + 8u * 1024 * 1024);     // 1 MB
  float*  logits = (float*)(ws + 9u * 1024 * 1024);      // 2 MB
  __bf16* W1b    = (__bf16*)(ws + 11u * 1024 * 1024);    // 128 KB
  __bf16* W2b    = (__bf16*)(ws + 11u * 1024 * 1024 + 256u * 1024);  // 32 KB

  // Pre-convert weights to bf16 (one time, tiny)
  cvt_bf16_kernel<<<64, 256, 0, stream>>>(W1, W1b);   // 256*256
  cvt_bf16_kernel<<<16, 256, 0, stream>>>(W2, W2b);   // 256*64
  // XW = x @ W1 (bf16)
  small_gemm_kernel<16, float><<<64, 256, 0, stream>>>(x, W1b, XW);
  // H = relu(adj @ XW) (bf16)   -- big pass 1 over adj, 256 blocks
  adj_gemm_kernel<16, true, __bf16><<<256, 256, 0, stream>>>(adj, XW, H);
  // HW2 = H @ W2 (bf16)
  small_gemm_kernel<4, __bf16><<<64, 256, 0, stream>>>(H, W2b, HW2);
  // logits = adj @ HW2 (f32)    -- big pass 2 over adj (K-reassociated: N=64)
  adj_gemm_kernel<4, false, float><<<256, 256, 0, stream>>>(adj, HW2, logits);
  // out = log_softmax(logits, axis=1)
  logsoftmax_kernel<<<GCN_N, 64, 0, stream>>>(logits, out);
}
